// FlashAttention_57784490000871
// MI455X (gfx1250) — matrix operand order repaired and checked by reading, not compiled, not run on hardware
//
#include <hip/hip_runtime.h>

typedef __attribute__((ext_vector_type(16))) _Float16 v16h;
typedef __attribute__((ext_vector_type(8)))  _Float16 v8h;
typedef __attribute__((ext_vector_type(4)))  _Float16 v4h;
typedef __attribute__((ext_vector_type(8)))  float    v8f;

union Frag { v16h v; v8h h[2]; };

#define N_CTX 2048
#define DHEAD 64
#define BR 64
#define BC 64
#define NBH  32                  // B*H = 2*16
#define TTOT (N_CTX / BC)        // 32 key tiles
#define LSTR 72                  // LDS row stride in halves (144B = 9*16B, bank pad)
#define ELEMS_PER_T ((size_t)NBH * N_CTX * DHEAD)   // 4,194,304 halves per tensor

// xor-shuffle within wave32 via ds_swizzle BITMASK_PERM: and=0x1f, or=0, xor=mask
#define XSWZ(v, mask) __builtin_bit_cast(float, \
    __builtin_amdgcn_ds_swizzle(__builtin_bit_cast(int, (v)), (0x1f | ((mask) << 10))))

// ---- async global->LDS copy of a 64-row x 64-half tile (8 KB), 4 x b128/thread ----
// ASYNCcnt-tracked (CDNA5 ISA); LDS offset = low 32 bits of
// the generic pointer (LDS aperture truncation, CDNA5 ISA).
__device__ __forceinline__ void async_tile(const _Float16* g, size_t rstride,
                                           _Float16 (*lds)[LSTR], int tid) {
#pragma unroll
    for (int j = 0; j < 4; ++j) {
        int c   = tid + j * 128;          // 16B chunk id, 0..511
        int r   = c >> 3;                 // row 0..63
        int col = (c & 7) << 3;           // half offset in row (multiple of 8)
        unsigned loff = (unsigned)(uintptr_t)&lds[r][col];
        const _Float16* gp = g + (size_t)r * rstride + col;
        asm volatile("global_load_async_to_lds_b128 %0, %1, off"
                     :: "v"(loff), "v"(gp) : "memory");
    }
}
__device__ __forceinline__ void wait_async0() {
    asm volatile("s_wait_asynccnt 0x0" ::: "memory");
}

// ================= pre-pass 1: f32 -> f16 straight convert (Q, K) =================
__global__ __launch_bounds__(256)
void cvt_f32_f16_kernel(const float* __restrict__ src, _Float16* __restrict__ dst) {
    size_t i4 = (size_t)blockIdx.x * 256 + threadIdx.x;   // float4 index
    float4 f = ((const float4*)src)[i4];
    v4h h = {(_Float16)f.x, (_Float16)f.y, (_Float16)f.z, (_Float16)f.w};
    *(v4h*)(dst + i4 * 4) = h;
}

// ====== pre-pass 2: V f32 [bh][N][d] -> f16 transposed [bh][d][N] via LDS tile ======
__global__ __launch_bounds__(256)
void transpose_v_kernel(const float* __restrict__ V, _Float16* __restrict__ Vt) {
    __shared__ _Float16 tile[64][LSTR];
    const int key0 = blockIdx.x * 64;
    const int bh   = blockIdx.y;
    const int tid  = threadIdx.x;
    const float* in = V + ((size_t)bh * N_CTX + key0) * DHEAD;
#pragma unroll
    for (int i = 0; i < 16; ++i) {
        int e = tid + i * 256;
        int r = e >> 6, c = e & 63;                 // r=key, c=d  (coalesced read)
        tile[r][c] = (_Float16)in[(size_t)r * DHEAD + c];
    }
    __syncthreads();
    _Float16* out = Vt + (size_t)bh * DHEAD * N_CTX + key0;
#pragma unroll
    for (int i = 0; i < 16; ++i) {
        int e = tid + i * 256;
        int d = e >> 6, k = e & 63;                 // coalesced write along keys
        out[(size_t)d * N_CTX + k] = tile[k][d];
    }
}

// ========================== flash attention forward ==========================
__global__ __launch_bounds__(128)
void FlashAttention_57784490000871_kernel(const _Float16* __restrict__ Qh,
                                          const _Float16* __restrict__ Kh,
                                          const _Float16* __restrict__ Vth,
                                          float* __restrict__ O)
{
    __shared__ _Float16 Qs[BR][LSTR];          // Q tile [row][d]
    __shared__ _Float16 Ks[2][BC][LSTR];       // K tiles, double buffered [key][d]
    __shared__ _Float16 Vs[2][DHEAD][LSTR];    // V^T tiles, double buffered [d][key]
    __shared__ _Float16 Ps[BR][LSTR];          // P staging [row][key]

    const int tid  = threadIdx.x;
    const int lane = tid & 31;
    const int wave = tid >> 5;      // 0..3
    const int hf   = lane >> 4;     // half-wave 0/1
    const int ln   = lane & 15;

    const int row_tile = blockIdx.x;   // 0..31
    const int bh       = blockIdx.y;   // 0..31

    const float kLogScale = 0.125f * 1.4426950408889634f; // (1/sqrt(64))*log2(e)

    const _Float16* Qg  = Qh  + ((size_t)bh * N_CTX + (size_t)row_tile * BR) * DHEAD;
    const _Float16* Kg  = Kh  + (size_t)bh * N_CTX * DHEAD;   // [key][d], rstride 64
    const _Float16* Vtg = Vth + (size_t)bh * DHEAD * N_CTX;   // [d][key], rstride 2048
    float*          Og  = O   + ((size_t)bh * N_CTX + (size_t)row_tile * BR) * DHEAD;

    // ---- prologue: async-stage Q and tile 0 of K, V^T ----
    async_tile(Qg,  DHEAD, Qs,    tid);
    async_tile(Kg,  DHEAD, Ks[0], tid);
    async_tile(Vtg, N_CTX, Vs[0], tid);
    wait_async0();
    __syncthreads();

    // ---- per-wave Q A-fragments (16-bit A 16x32 layout) ----
    Frag qf[2];
    const int qrow = wave * 16 + ln;
#pragma unroll
    for (int kc = 0; kc < 2; ++kc) {
        qf[kc].h[0] = *(const v8h*)&Qs[qrow][kc * 32 + hf * 8];
        qf[kc].h[1] = *(const v8h*)&Qs[qrow][kc * 32 + 16 + hf * 8];
    }

    v8f oacc[4];
#pragma unroll
    for (int nt = 0; nt < 4; ++nt) oacc[nt] = (v8f){};
    float m[8], l[8];
#pragma unroll
    for (int r = 0; r < 8; ++r) { m[r] = -1.0e30f; l[r] = 0.0f; }

    for (int t = 0; t < TTOT; ++t) {
        const int buf = t & 1;

        // stream next tile into the other buffer while we compute on this one
        if (t + 1 < TTOT) {
            async_tile(Kg  + (size_t)(t + 1) * BC * DHEAD, DHEAD, Ks[buf ^ 1], tid);
            async_tile(Vtg + (size_t)(t + 1) * BC,         N_CTX, Vs[buf ^ 1], tid);
        }

        // ---- S = Q * K^T : 16x64 per wave, 8 WMMAs ----
        v8f s[4];
#pragma unroll
        for (int nt = 0; nt < 4; ++nt) {
            v8f acc = (v8f){};
#pragma unroll
            for (int kc = 0; kc < 2; ++kc) {
                Frag bf;
                const _Float16* kp = &Ks[buf][nt * 16 + ln][kc * 32 + hf * 8];
                bf.h[0] = *(const v8h*)kp;
                bf.h[1] = *(const v8h*)(kp + 16);
                acc = __builtin_amdgcn_wmma_f32_16x16x32_f16(
                        false, qf[kc].v, false, bf.v, (short)0, acc, false, false);
            }
            s[nt] = acc * kLogScale;   // vector scale into log2 domain
        }

        // ---- online softmax per accumulator row ----
#pragma unroll
        for (int r = 0; r < 8; ++r) {
            float t0 = s[0][r], t1 = s[1][r], t2 = s[2][r], t3 = s[3][r];
            float mv = fmaxf(fmaxf(t0, t1), fmaxf(t2, t3));
            mv = fmaxf(mv, XSWZ(mv, 1));
            mv = fmaxf(mv, XSWZ(mv, 2));
            mv = fmaxf(mv, XSWZ(mv, 4));
            mv = fmaxf(mv, XSWZ(mv, 8));
            float mnew  = fmaxf(m[r], mv);
            float alpha = __builtin_amdgcn_exp2f(m[r] - mnew);
            m[r] = mnew;
            float p0 = __builtin_amdgcn_exp2f(t0 - mnew);
            float p1 = __builtin_amdgcn_exp2f(t1 - mnew);
            float p2 = __builtin_amdgcn_exp2f(t2 - mnew);
            float p3 = __builtin_amdgcn_exp2f(t3 - mnew);
            s[0][r] = p0; s[1][r] = p1; s[2][r] = p2; s[3][r] = p3;
            float ps = (p0 + p1) + (p2 + p3);
            ps += XSWZ(ps, 1);
            ps += XSWZ(ps, 2);
            ps += XSWZ(ps, 4);
            ps += XSWZ(ps, 8);
            l[r] = l[r] * alpha + ps;
            oacc[0][r] *= alpha;
            oacc[1][r] *= alpha;
            oacc[2][r] *= alpha;
            oacc[3][r] *= alpha;
        }

        // ---- stage P to per-wave LDS stripe (C-layout -> A-layout) ----
        const int prow_base = wave * 16 + 8 * hf;
#pragma unroll
        for (int nt = 0; nt < 4; ++nt)
#pragma unroll
            for (int r = 0; r < 8; ++r)
                Ps[prow_base + r][nt * 16 + ln] = (_Float16)s[nt][r];
        // per-wave rows only; same-wave LDS ops are in-order

        // ---- O += P * V : 8 WMMAs ----
        const int prow = wave * 16 + ln;
#pragma unroll
        for (int kc = 0; kc < 2; ++kc) {
            Frag af;
            af.h[0] = *(const v8h*)&Ps[prow][kc * 32 + hf * 8];
            af.h[1] = *(const v8h*)&Ps[prow][kc * 32 + 16 + hf * 8];
#pragma unroll
            for (int nt = 0; nt < 4; ++nt) {
                Frag bf;
                const _Float16* vp = &Vs[buf][nt * 16 + ln][kc * 32 + hf * 8];
                bf.h[0] = *(const v8h*)vp;
                bf.h[1] = *(const v8h*)(vp + 16);
                oacc[nt] = __builtin_amdgcn_wmma_f32_16x16x32_f16(
                        false, af.v, false, bf.v, (short)0, oacc[nt], false, false);
            }
        }

        // my next-tile asyncs done + all waves finished reading buf before reuse
        wait_async0();
        __syncthreads();
    }

    // ---- normalize and store O (fp32) ----
#pragma unroll
    for (int r = 0; r < 8; ++r) {
        float inv = 1.0f / l[r];
        int row = wave * 16 + 8 * hf + r;
#pragma unroll
        for (int nt = 0; nt < 4; ++nt)
            Og[row * DHEAD + nt * 16 + ln] = oacc[nt][r] * inv;
    }
}

extern "C" void kernel_launch(void* const* d_in, const int* in_sizes, int n_in,
                              void* d_out, int out_size, void* d_ws, size_t ws_size,
                              hipStream_t stream) {
    const float* Q = (const float*)d_in[0];
    const float* K = (const float*)d_in[1];
    const float* V = (const float*)d_in[2];
    float* O = (float*)d_out;

    // workspace layout (f16): Qh | Kh | Vt  -> 3 * 8 MiB = 25,165,824 bytes
    _Float16* Qh = (_Float16*)d_ws;
    _Float16* Kh = Qh + ELEMS_PER_T;
    _Float16* Vt = Kh + ELEMS_PER_T;

    const int cvt_blocks = (int)(ELEMS_PER_T / 4 / 256);   // 4096
    cvt_f32_f16_kernel<<<cvt_blocks, 256, 0, stream>>>(Q, Qh);
    cvt_f32_f16_kernel<<<cvt_blocks, 256, 0, stream>>>(K, Kh);
    transpose_v_kernel<<<dim3(N_CTX / 64, NBH), 256, 0, stream>>>(V, Vt);

    FlashAttention_57784490000871_kernel<<<dim3(N_CTX / BR, NBH), 128, 0, stream>>>(
        Qh, Kh, Vt, O);
}
